// MultiHeadCrossAttentionLayer_26302379720810
// MI455X (gfx1250) — matrix operand order repaired and checked by reading, not compiled, not run on hardware
//
#include <hip/hip_runtime.h>

// ---------------------------------------------------------------- types
typedef __attribute__((ext_vector_type(16))) __bf16 v16bf;
typedef __attribute__((ext_vector_type(8)))  __bf16 bf16x8;
typedef __attribute__((ext_vector_type(8)))  float  v8f;
typedef __attribute__((ext_vector_type(4)))  float  f32x4;

#define DIM    768
#define NHEAD  12
#define HDIM   64
#define BATCH  8
#define QLEN   512
#define KVLEN  4096
#define KVW    (KVLEN / 32)      // mask words per row
#define LOG2E  1.4426950408889634f

#if __has_builtin(__builtin_amdgcn_exp2f)
#define EXP2F(x) __builtin_amdgcn_exp2f(x)
#else
#define EXP2F(x) exp2f(x)
#endif

// ---------------------------------------------------------------- WMMA
static __device__ inline v8f wmma_bf16(v16bf a, v16bf b, v8f c) {
    return __builtin_amdgcn_wmma_f32_16x16x32_bf16(
        /*neg_a=*/false, a, /*neg_b=*/false, b,
        /*c_mod=*/(short)0, c, /*reuse_a=*/false, /*reuse_b=*/false);
}

// A-fragment 16x32 bf16 from f32 row (contiguous layout:
// elems 0..15 -> K = half*16 + e)
static __device__ inline v16bf a_frag_from_f32(const float* __restrict__ rowp,
                                               int kb, int half) {
    const f32x4* p0 = (const f32x4*)(rowp + kb + half * 16);
    const f32x4* p1 = (const f32x4*)(rowp + kb + half * 16 + 8);
    f32x4 x0 = p0[0], x1 = p0[1], x2 = p1[0], x3 = p1[1];
    v16bf r;
#pragma unroll
    for (int j = 0; j < 4; ++j) {
        r[j]      = (__bf16)x0[j];
        r[4 + j]  = (__bf16)x1[j];
        r[8 + j]  = (__bf16)x2[j];
        r[12 + j] = (__bf16)x3[j];
    }
    return r;
}

// B-fragment 32x16 bf16 from f32 row (lane = column n, K = half*16 + e)
static __device__ inline v16bf b_frag_from_f32(const float* __restrict__ rowp,
                                               int kb, int half) {
    const f32x4* p = (const f32x4*)(rowp + kb + half * 16);
    f32x4 x0 = p[0], x1 = p[1], x2 = p[2], x3 = p[3];
    v16bf r;
#pragma unroll
    for (int j = 0; j < 4; ++j) {
        r[j]      = (__bf16)x0[j];
        r[4 + j]  = (__bf16)x1[j];
        r[8 + j]  = (__bf16)x2[j];
        r[12 + j] = (__bf16)x3[j];
    }
    return r;
}

// A-fragment directly from bf16 memory (global or LDS)
static __device__ inline v16bf a_frag_bf16(const __bf16* p, int half) {
    union { v16bf v; bf16x8 h[2]; } u;
    u.h[0] = *(const bf16x8*)(p + half * 16);
    u.h[1] = *(const bf16x8*)(p + half * 16 + 8);
    return u.v;
}

// B-fragment: 16 contiguous bf16 (caller applies the half*16 offset)
static __device__ inline v16bf frag_contig16(const __bf16* p) {
    union { v16bf v; bf16x8 h[2]; } u;
    u.h[0] = *(const bf16x8*)(p);
    u.h[1] = *(const bf16x8*)(p + 8);
    return u.v;
}

// ---------------------------------------------------------------- LayerNorm
__global__ void ln_kernel(const float* __restrict__ x,
                          const float* __restrict__ g,
                          const float* __restrict__ bta,
                          float* __restrict__ out) {
    int row = blockIdx.x;                       // 0..4095
    const float* xr = x + (size_t)row * DIM;
    float v[3], s = 0.f, s2 = 0.f;
#pragma unroll
    for (int i = 0; i < 3; ++i) {
        v[i] = xr[threadIdx.x + i * 256];
        s += v[i]; s2 += v[i] * v[i];
    }
#pragma unroll
    for (int m = 1; m < 32; m <<= 1) {
        s  += __shfl_xor(s,  m, 32);
        s2 += __shfl_xor(s2, m, 32);
    }
    __shared__ float ps[8], ps2[8];
    int w = threadIdx.x >> 5, l = threadIdx.x & 31;
    if (l == 0) { ps[w] = s; ps2[w] = s2; }
    __syncthreads();
    if (w == 0) {
        float a = ps[l & 7], b2 = ps2[l & 7];
#pragma unroll
        for (int m = 1; m < 8; m <<= 1) {
            a  += __shfl_xor(a,  m, 32);
            b2 += __shfl_xor(b2, m, 32);
        }
        if (l == 0) { ps[0] = a; ps2[0] = b2; }
    }
    __syncthreads();
    float mu  = ps[0] * (1.f / DIM);
    float var = ps2[0] * (1.f / DIM) - mu * mu;
    float r   = rsqrtf(var + 1e-5f);
#pragma unroll
    for (int i = 0; i < 3; ++i) {
        int c = threadIdx.x + i * 256;
        out[(size_t)row * DIM + c] = (v[i] - mu) * r * g[c] + bta[c];
    }
}

// ---------------------------------------------------------------- mask pack
// One wave -> one 32-bit word: bit j = (mask[...]==1) for 32 consecutive kv.
__global__ void pack_mask(const int* __restrict__ mask,
                          unsigned int* __restrict__ mpack) {
    int lane = threadIdx.x & 31;
    size_t gw = (size_t)blockIdx.x * (blockDim.x >> 5) + (threadIdx.x >> 5);
    int v = mask[gw * 32 + lane];
    unsigned int bal = __builtin_amdgcn_ballot_w32(v == 1);
    if (lane == 0) mpack[gw] = bal;
}

// ---------------------------------------------------------------- GEMM
// C[M,768] = (A[M,768] @ W[768,768]^T + bias) * scale
// OM: 0 = f32 row-major, 1 = bf16 row-major, 2 = bf16 transposed V layout
// block = 256 threads (8 waves), wave tile 32x32, block tile 128(M) x 64(N)
template <int OM>
__global__ void gemm768(const float* __restrict__ A,
                        const float* __restrict__ W,
                        const float* __restrict__ bias,
                        void* __restrict__ outp, float scale) {
    const int K = DIM, N = DIM;
    int lane = threadIdx.x & 31, wv = threadIdx.x >> 5;
    int wm = wv & 3, wn = wv >> 2;
    int half = lane >> 4, lq = lane & 15;
    int m_base = blockIdx.x * 128 + wm * 32;
    int n_base = blockIdx.y * 64 + wn * 32;

    const float* arow[2] = { A + (size_t)(m_base + lq) * K,
                             A + (size_t)(m_base + 16 + lq) * K };
    const float* brow[2] = { W + (size_t)(n_base + lq) * K,
                             W + (size_t)(n_base + 16 + lq) * K };
    v8f acc[2][2] = {};

#pragma unroll 4
    for (int kb = 0; kb < K; kb += 32) {
        __builtin_prefetch(arow[0] + kb + 128, 0, 1);
        __builtin_prefetch(brow[0] + kb + 128, 0, 1);
        v16bf af[2], bf_[2];
        af[0]  = a_frag_from_f32(arow[0], kb, half);
        af[1]  = a_frag_from_f32(arow[1], kb, half);
        bf_[0] = b_frag_from_f32(brow[0], kb, half);
        bf_[1] = b_frag_from_f32(brow[1], kb, half);
#pragma unroll
        for (int mt = 0; mt < 2; ++mt)
#pragma unroll
            for (int nt = 0; nt < 2; ++nt)
                acc[mt][nt] = wmma_bf16(af[mt], bf_[nt], acc[mt][nt]);
    }

#pragma unroll
    for (int nt = 0; nt < 2; ++nt) {
        int n = n_base + nt * 16 + lq;
        float bv = bias[n];
#pragma unroll
        for (int mt = 0; mt < 2; ++mt) {
#pragma unroll
            for (int r = 0; r < 8; ++r) {
                int m = m_base + mt * 16 + half * 8 + r;
                float val = (acc[mt][nt][r] + bv) * scale;
                if (OM == 0) {
                    ((float*)outp)[(size_t)m * N + n] = val;
                } else if (OM == 1) {
                    ((__bf16*)outp)[(size_t)m * N + n] = (__bf16)val;
                } else {
                    int bb = m >> 12, kv = m & (KVLEN - 1);
                    int h = n >> 6, d = n & (HDIM - 1);
                    ((__bf16*)outp)[((size_t)((bb * NHEAD + h) * HDIM + d)) * KVLEN + kv]
                        = (__bf16)val;
                }
            }
        }
    }
}

// ---------------------------------------------------------------- Flash attention
// grid (B*H, QLEN/128), 256 threads = 8 waves.
// Wave tile: 16 q-rows x 64 kv per iteration. All global accesses in the kv
// loop use ONE running base pointer each (K, V, mask) with compile-time
// instruction offsets, so no per-tile 64-bit address arithmetic / spills.
__global__ void attn_kernel(const __bf16* __restrict__ Qp,   // [B,Q,768] (pre-scaled)
                            const __bf16* __restrict__ Kp,   // [B,KV,768]
                            const __bf16* __restrict__ Vt,   // [B,H,64,KV]
                            const unsigned int* __restrict__ mpack, // [B,Q,KV/32]
                            float*        __restrict__ attn_out) {  // [B,Q,768]
    int b = blockIdx.x / NHEAD, h = blockIdx.x % NHEAD;
    int lane = threadIdx.x & 31, wv = threadIdx.x >> 5;
    int half = lane >> 4, lq = lane & 15;
    int q0 = blockIdx.y * 128 + wv * 16;

    __shared__ __align__(16) __bf16 pbuf[8][16][64];  // P staging (per wave)
    __shared__ __align__(16) __bf16 qbuf[8][16][HDIM];// Q tile (per wave)
    __bf16(*pw)[64]   = pbuf[wv];
    __bf16(*qw)[HDIM] = qbuf[wv];

    // ---- stage this wave's Q tile (16 rows x 64) into LDS:
    // lane L -> row L&15, 32-col half (L>>4)*32
    {
        const bf16x8* src = (const bf16x8*)(Qp
            + ((size_t)(b * QLEN + q0 + lq)) * DIM + h * HDIM + half * 32);
        bf16x8* dst = (bf16x8*)&qw[lq][half * 32];
        dst[0] = src[0];
        dst[1] = src[1];
    }
    asm volatile("s_wait_dscnt 0" ::: "memory");

    float mstat[8], lstat[8];
#pragma unroll
    for (int r = 0; r < 8; ++r) { mstat[r] = -1e30f; lstat[r] = 0.f; }
    v8f o[4] = {};

    // Running base pointers (advanced once per kv tile):
    const __bf16* kptr = Kp + ((size_t)(b * KVLEN) + lq) * DIM + h * HDIM + half * 16;
    const __bf16* vptr = Vt + ((size_t)((b * NHEAD + h) * HDIM) + lq) * KVLEN + half * 16;
    const unsigned int* mptr = mpack + (size_t)(b * QLEN + q0 + half * 8) * KVW;

#pragma unroll 1
    for (int kv0 = 0; kv0 < KVLEN; kv0 += 64) {
        // ---- S = Q @ K^T  (scale folded into Q); Q frags from LDS.
        // K offsets: nt*16 rows (nt*16*DIM elems) + ks*32 cols -> constants.
        v8f s[4] = {};
#pragma unroll
        for (int ks = 0; ks < 2; ++ks) {
            v16bf qa = a_frag_bf16(&qw[lq][ks * 32], half);
            v16bf kf[4];
#pragma unroll
            for (int nt = 0; nt < 4; ++nt)
                kf[nt] = frag_contig16(kptr + nt * (16 * DIM) + ks * 32);
#pragma unroll
            for (int nt = 0; nt < 4; ++nt)
                s[nt] = wmma_bf16(qa, kf[nt], s[nt]);
        }
        // ---- mask via packed bits: 16 independent 4B loads (const offsets)
        unsigned int wbits[2][8];
#pragma unroll
        for (int w = 0; w < 2; ++w)
#pragma unroll
            for (int r = 0; r < 8; ++r)
                wbits[w][r] = mptr[r * KVW + w];
#pragma unroll
        for (int nt = 0; nt < 4; ++nt) {
            int sh = (nt & 1) * 16 + lq;
#pragma unroll
            for (int r = 0; r < 8; ++r)
                if ((wbits[nt >> 1][r] >> sh) & 1u) s[nt][r] = -1e30f;
        }
        // ---- online softmax (rows live in 16-lane halves)
#pragma unroll
        for (int r = 0; r < 8; ++r) {
            float rm = fmaxf(fmaxf(s[0][r], s[1][r]), fmaxf(s[2][r], s[3][r]));
#pragma unroll
            for (int xm = 1; xm < 16; xm <<= 1)
                rm = fmaxf(rm, __shfl_xor(rm, xm, 32));
            float mold = mstat[r];
            float mnew = fmaxf(mold, rm);
            float alpha = EXP2F((mold - mnew) * LOG2E);
            float p0 = EXP2F((s[0][r] - mnew) * LOG2E);
            float p1 = EXP2F((s[1][r] - mnew) * LOG2E);
            float p2 = EXP2F((s[2][r] - mnew) * LOG2E);
            float p3 = EXP2F((s[3][r] - mnew) * LOG2E);
            s[0][r] = p0; s[1][r] = p1; s[2][r] = p2; s[3][r] = p3;
            float rs = (p0 + p1) + (p2 + p3);
#pragma unroll
            for (int xm = 1; xm < 16; xm <<= 1)
                rs += __shfl_xor(rs, xm, 32);
            lstat[r] = lstat[r] * alpha + rs;
            mstat[r] = mnew;
#pragma unroll
            for (int nt2 = 0; nt2 < 4; ++nt2) o[nt2][r] *= alpha;
        }
        // ---- stage P (16x64) into LDS, reload as two A-fragments (k=0..63)
#pragma unroll
        for (int nt = 0; nt < 4; ++nt)
#pragma unroll
            for (int r = 0; r < 8; ++r)
                pw[half * 8 + r][nt * 16 + lq] = (__bf16)s[nt][r];
        asm volatile("s_wait_dscnt 0" ::: "memory");
        v16bf pa[2];
        pa[0] = a_frag_bf16(&pw[lq][0], half);
        pa[1] = a_frag_bf16(&pw[lq][32], half);
        // ---- O += P @ V  (Vt transposed: nt2*16*KVLEN + ks2*32 = constants)
#pragma unroll
        for (int ks2 = 0; ks2 < 2; ++ks2) {
            v16bf vb[4];
#pragma unroll
            for (int nt2 = 0; nt2 < 4; ++nt2)
                vb[nt2] = frag_contig16(vptr + nt2 * (16 * KVLEN) + ks2 * 32);
#pragma unroll
            for (int nt2 = 0; nt2 < 4; ++nt2)
                o[nt2] = wmma_bf16(pa[ks2], vb[nt2], o[nt2]);
        }
        kptr += 64 * DIM;   // next 64 kv rows
        vptr += 64;         // next 64 kv columns
        mptr += 2;          // next 2 mask words
    }

    // ---- epilogue: O /= l, merge heads into [B,Q,768] f32
#pragma unroll
    for (int r = 0; r < 8; ++r) {
        float inv = 1.f / lstat[r];
        int m = q0 + half * 8 + r;
#pragma unroll
        for (int nt2 = 0; nt2 < 4; ++nt2)
            attn_out[((size_t)(b * QLEN + m)) * DIM + h * HDIM + nt2 * 16 + lq]
                = o[nt2][r] * inv;
    }
}

// ---------------------------------------------------------------- launcher
extern "C" void kernel_launch(void* const* d_in, const int* in_sizes, int n_in,
                              void* d_out, int out_size, void* d_ws, size_t ws_size,
                              hipStream_t stream) {
    const float* query = (const float*)d_in[0];
    const float* key_i = (const float*)d_in[1];
    const float* val_i = (const float*)d_in[2];
    const int*   amask = (const int*)d_in[3];
    const float* Wq = (const float*)d_in[4];  const float* bq = (const float*)d_in[5];
    const float* Wk = (const float*)d_in[6];  const float* bk = (const float*)d_in[7];
    const float* Wv = (const float*)d_in[8];  const float* bv = (const float*)d_in[9];
    const float* Wo = (const float*)d_in[10]; const float* bo = (const float*)d_in[11];
    const float* lng = (const float*)d_in[12]; const float* lnb = (const float*)d_in[13];

    char* ws = (char*)d_ws;
    float*  qn    = (float*)ws;        ws += (size_t)BATCH * QLEN * DIM * 4;  // 12.6 MB
    __bf16* Qp    = (__bf16*)ws;       ws += (size_t)BATCH * QLEN * DIM * 2;  //  6.3 MB
    __bf16* Kp    = (__bf16*)ws;       ws += (size_t)BATCH * KVLEN * DIM * 2; // 50.3 MB
    __bf16* Vt    = (__bf16*)ws;       ws += (size_t)BATCH * KVLEN * DIM * 2; // 50.3 MB
    float*  attn  = (float*)ws;        ws += (size_t)BATCH * QLEN * DIM * 4;  // 12.6 MB
    unsigned int* mpk = (unsigned int*)ws;                                    //  2.0 MB

    const float qscale = 0.125f;  // HEAD_DIM^-0.5 folded into Q projection

    ln_kernel<<<BATCH * QLEN, 256, 0, stream>>>(query, lng, lnb, qn);
    // 524288 mask words, 8 words per 256-thread block
    pack_mask<<<(BATCH * QLEN * KVW) / 8, 256, 0, stream>>>(amask, mpk);
    gemm768<1><<<dim3((BATCH * QLEN) / 128, DIM / 64), 256, 0, stream>>>(
        qn, Wq, bq, (void*)Qp, qscale);
    gemm768<1><<<dim3((BATCH * KVLEN) / 128, DIM / 64), 256, 0, stream>>>(
        key_i, Wk, bk, (void*)Kp, 1.f);
    gemm768<2><<<dim3((BATCH * KVLEN) / 128, DIM / 64), 256, 0, stream>>>(
        val_i, Wv, bv, (void*)Vt, 1.f);
    attn_kernel<<<dim3(BATCH * NHEAD, QLEN / 128), 256, 0, stream>>>(
        Qp, Kp, Vt, mpk, attn);
    gemm768<0><<<dim3((BATCH * QLEN) / 128, DIM / 64), 256, 0, stream>>>(
        attn, Wo, bo, d_out, 1.f);
}
